// MOE_9371618639913
// MI455X (gfx1250) — matrix operand order repaired and checked by reading, not compiled, not run on hardware
//
#include <hip/hip_runtime.h>
#include <hip/hip_bf16.h>

// ---------------------------------------------------------------------------
// MoE forward for MI455X (gfx1250), wave32, WMMA bf16 16x16x32.
// B=4,S=2048 -> T=8192 tokens, D=1024, F=4096, E=8, TOP_K=2.
// Fast path: weights pre-converted to bf16 packed in WMMA B-fragment order
// (two coalesced global_load_b128 per fragment, zero staging in hot loop).
// ---------------------------------------------------------------------------

#define T_TOK 8192
#define D_DIM 1024
#define F_DIM 4096
#define E_NUM 8
#define FC    128          // F-chunk per iteration (8 waves x 16 cols)
#define XS    1032         // lds_x row stride in halfs (2064B, 16B aligned)
#define HS    136          // lds_h row stride in halfs (272B)
#define WSLOT 40           // fallback weight slot row stride in halfs

typedef __bf16 bf16_t;
typedef __attribute__((ext_vector_type(16))) __bf16 v16bf;
typedef __attribute__((ext_vector_type(8)))  __bf16 v8bf;
typedef __attribute__((ext_vector_type(4)))  __bf16 v4bf;
typedef __attribute__((ext_vector_type(8)))  float  v8f;

__device__ __forceinline__ v16bf make_frag(const bf16_t* p0, const bf16_t* p1) {
  v8bf lo = *(const v8bf*)p0;   // ds_load_b128 / global_load_b128
  v8bf hi = *(const v8bf*)p1;
  return __builtin_shufflevector(lo, hi, 0,1,2,3,4,5,6,7,8,9,10,11,12,13,14,15);
}

__device__ __forceinline__ v8f wmma_bf16(v16bf a, v16bf b, v8f c) {
  return __builtin_amdgcn_wmma_f32_16x16x32_bf16(
      false, a, false, b, (short)0, c, false, false);
}

// silu without the IEEE divide chain: u * rcp(1+exp(-u))  (v_rcp_f32)
__device__ __forceinline__ float silu_fast(float u) {
  return u * __builtin_amdgcn_rcpf(1.0f + __expf(-u));
}

// ---------------------------------------------------------------------------
// Kernel 0: zero output accumulation region + workspace header
// ---------------------------------------------------------------------------
__global__ void moe_init_kernel(float* __restrict__ out, size_t n,
                                unsigned int* __restrict__ hdr) {
  size_t i = (size_t)blockIdx.x * blockDim.x + threadIdx.x;
  const size_t stride = (size_t)gridDim.x * blockDim.x;
  for (; i < n; i += stride) out[i] = 0.0f;
  if (blockIdx.x == 0 && threadIdx.x < 32) hdr[threadIdx.x] = 0u;
}

// ---------------------------------------------------------------------------
// Kernel P: pack a [E][rowsK][cols] f32 weight tensor into bf16 tiles laid out
// in WMMA B-fragment order. Tile = 32 K-rows x 16 N-cols. Within a tile,
// lane l (0..31) owns column n=l&15, K-run [16*(l>>4), +16) -> 16 contiguous
// halfs at dst[tile*512 + l*16]. A wave then fetches a whole B fragment as
// 1KB of contiguous, coalesced global memory.
// ---------------------------------------------------------------------------
__global__ __launch_bounds__(256) void moe_pack_kernel(
    const float* __restrict__ src, bf16_t* __restrict__ dst,
    int rowsK, int cols) {
  const int ksteps = rowsK >> 5;           // tiles along K
  const int ctiles = cols  >> 4;           // tiles along N
  const size_t gid = (size_t)blockIdx.x * 256 + threadIdx.x;
  const int lane = (int)(gid & 31);
  const size_t tile = gid >> 5;
  const int ks = (int)(tile % ksteps);
  const int ct = (int)((tile / ksteps) % ctiles);
  const int e  = (int)(tile / ((size_t)ksteps * ctiles));
  const int n  = lane & 15;
  const int hi = lane >> 4;

  const float* s = src + ((size_t)e * rowsK + ks * 32 + 16 * hi) * cols
                       + ct * 16 + n;
  v16bf v;
#pragma unroll
  for (int j = 0; j < 16; j++) v[j] = (bf16_t)s[(size_t)j * cols];

  bf16_t* d = dst + gid * 16;
  *(v8bf*)d = __builtin_shufflevector(v, v, 0,1,2,3,4,5,6,7);
  *(v8bf*)(d + 8) = __builtin_shufflevector(v, v, 8,9,10,11,12,13,14,15);
}

// ---------------------------------------------------------------------------
// Kernel 1: router — logits, z/p stats, top-2 gates, per-expert token lists.
// One wave per token; 8 waves per block.
// ---------------------------------------------------------------------------
__global__ __launch_bounds__(256) void moe_router_kernel(
    const float* __restrict__ x, const float* __restrict__ Wr,
    unsigned int* __restrict__ cnt, float* __restrict__ z_sum,
    float* __restrict__ p_sum, int* __restrict__ toklist,
    float* __restrict__ gatelist) {
  const int lane = threadIdx.x & 31;
  const int wv   = threadIdx.x >> 5;
  const int t    = blockIdx.x * 8 + wv;
  if (t >= T_TOK) return;

  float acc[E_NUM];
#pragma unroll
  for (int e = 0; e < E_NUM; e++) acc[e] = 0.0f;

  const float* xr = x + (size_t)t * D_DIM;
  for (int d = lane; d < D_DIM; d += 32) {
    const float xv = xr[d];
    const float* wr = Wr + (size_t)d * E_NUM;
#pragma unroll
    for (int e = 0; e < E_NUM; e++) acc[e] += xv * wr[e];
  }
#pragma unroll
  for (int e = 0; e < E_NUM; e++) {
    for (int off = 16; off > 0; off >>= 1) acc[e] += __shfl_xor(acc[e], off, 32);
  }

  if (lane == 0) {
    float mx = acc[0]; int i0 = 0;
#pragma unroll
    for (int e = 1; e < E_NUM; e++) if (acc[e] > mx) { mx = acc[e]; i0 = e; }
    float se = 0.0f;
#pragma unroll
    for (int e = 0; e < E_NUM; e++) se += __expf(acc[e] - mx);
    const float lse = mx + __logf(se);
    atomicAdd(z_sum, lse * lse);
#pragma unroll
    for (int e = 0; e < E_NUM; e++) atomicAdd(p_sum + e, __expf(acc[e] - lse));

    float mx2 = -3.4e38f; int i1 = (i0 == 0) ? 1 : 0;
#pragma unroll
    for (int e = 0; e < E_NUM; e++)
      if (e != i0 && acc[e] > mx2) { mx2 = acc[e]; i1 = e; }

    const float g0 = 1.0f / (1.0f + __expf(mx2 - mx));
    const float g1 = 1.0f - g0;

    unsigned p0 = atomicAdd(cnt + i0, 1u);
    toklist [(size_t)i0 * T_TOK + p0] = t;
    gatelist[(size_t)i0 * T_TOK + p0] = g0;
    unsigned p1 = atomicAdd(cnt + i1, 1u);
    toklist [(size_t)i1 * T_TOK + p1] = t;
    gatelist[(size_t)i1 * T_TOK + p1] = g1;
  }
}

// ---------------------------------------------------------------------------
// Kernel 2: finalize auxiliary outputs
// ---------------------------------------------------------------------------
__global__ void moe_finalize_kernel(const unsigned int* __restrict__ cnt,
                                    const float* __restrict__ z_sum,
                                    const float* __restrict__ p_sum,
                                    float* __restrict__ out_tail) {
  if (threadIdx.x == 0) {
    float lb = 0.0f;
#pragma unroll
    for (int e = 0; e < E_NUM; e++) {
      const float f = (float)cnt[e] / (float)(T_TOK * 2);
      out_tail[e] = f;
      lb += (p_sum[e] / (float)T_TOK) * f;
    }
    lb *= (float)E_NUM;
    const float z = z_sum[0] / (float)T_TOK;
    out_tail[8]  = z;
    out_tail[9]  = z * 0.001f;
    out_tail[10] = lb;
    out_tail[11] = lb * 0.1f;
  }
}

// ---------------------------------------------------------------------------
// Kernel 3 (FAST): fused expert MLP, M=32 tokens per block, packed bf16
// weights. 8 waves: wave w owns 16 f-cols of up/gate per 128-wide F chunk
// and 128 D-cols of the down accumulator, for both 16-row m-tiles.
// ---------------------------------------------------------------------------
__global__ __launch_bounds__(256) void moe_expert_fast_kernel(
    const float* __restrict__ x,
    const bf16_t* __restrict__ pkUp,    // packed [E][F/16][D/32][32][16]
    const bf16_t* __restrict__ pkGate,
    const bf16_t* __restrict__ pkDn,    // packed [E][D/16][F/32][32][16]
    const unsigned int* __restrict__ cnt,
    const int* __restrict__ toklist,
    const float* __restrict__ gatelist,
    float* __restrict__ out) {
  const int e    = blockIdx.y;
  const unsigned n_e = cnt[e];
  const int row0 = blockIdx.x * 32;
  if ((unsigned)row0 >= n_e) return;

  const int tid  = threadIdx.x;
  const int lane = tid & 31;
  const int w    = tid >> 5;
  const int hi   = lane >> 4;
  const int ln   = lane & 15;

  __shared__ int    s_tok[32];
  __shared__ float  s_gate[32];
  __shared__ __align__(16) bf16_t lds_x[32 * XS];
  __shared__ __align__(16) bf16_t lds_h[32 * HS];

  if (tid < 32) {
    const int r = row0 + tid;
    const bool v = (unsigned)r < n_e;
    s_tok[tid]  = v ? toklist [(size_t)e * T_TOK + r] : -1;
    s_gate[tid] = v ? gatelist[(size_t)e * T_TOK + r] : 0.0f;
  }
  __syncthreads();

  // Stage x tile (32 x 1024) f32 -> bf16 LDS.
  for (int idx = tid; idx < 32 * (D_DIM / 4); idx += 256) {
    const int m  = idx >> 8;
    const int d4 = (idx & 255) << 2;
    int t = s_tok[m]; if (t < 0) t = 0;
    const float4 v = *(const float4*)(x + (size_t)t * D_DIM + d4);
    v4bf b; b[0] = (bf16_t)v.x; b[1] = (bf16_t)v.y;
            b[2] = (bf16_t)v.z; b[3] = (bf16_t)v.w;
    *(v4bf*)(lds_x + m * XS + d4) = b;
  }
  __syncthreads();

  // Per-expert packed bases (tile = 512 halfs).
  const bf16_t* wuP = pkUp   + (size_t)e * (F_DIM / 16) * (D_DIM / 32) * 512;
  const bf16_t* wgP = pkGate + (size_t)e * (F_DIM / 16) * (D_DIM / 32) * 512;
  const bf16_t* wdP = pkDn   + (size_t)e * (D_DIM / 16) * (F_DIM / 32) * 512;

  v8f acc[2][8];
#pragma unroll
  for (int mt = 0; mt < 2; mt++)
#pragma unroll
    for (int i = 0; i < 8; i++) acc[mt][i] = v8f{};

  for (int fc = 0; fc < F_DIM; fc += FC) {
    const int nb = fc + w * 16;             // this wave's up/gate columns
    v8f uacc[2] = {v8f{}, v8f{}};
    v8f gacc[2] = {v8f{}, v8f{}};

    // ---- up & gate: [32 x 1024] @ [1024 x 16] ----
    const bf16_t* pu = wuP + ((size_t)(nb >> 4) * (D_DIM / 32)) * 512 + lane * 16;
    const bf16_t* pg = wgP + ((size_t)(nb >> 4) * (D_DIM / 32)) * 512 + lane * 16;
    for (int kk = 0; kk < D_DIM; kk += 32, pu += 512, pg += 512) {
      __builtin_prefetch(pu + 512, 0, 1);   // global_prefetch_b8
      const v16bf bu = make_frag(pu, pu + 8);
      const v16bf bg = make_frag(pg, pg + 8);
#pragma unroll
      for (int mt = 0; mt < 2; mt++) {
        const bf16_t* xr = lds_x + (mt * 16 + ln) * XS + kk;
        const v16bf a = make_frag(xr + 16 * hi, xr + 16 * hi + 8);
        uacc[mt] = wmma_bf16(a, bu, uacc[mt]);
        gacc[mt] = wmma_bf16(a, bg, gacc[mt]);
      }
    }

    // ---- h = silu(u) * g -> lds_h (bf16) ----
#pragma unroll
    for (int mt = 0; mt < 2; mt++)
#pragma unroll
      for (int v = 0; v < 8; v++) {
        const float hv = silu_fast(uacc[mt][v]) * gacc[mt][v];
        lds_h[(mt * 16 + v + 8 * hi) * HS + w * 16 + ln] = (bf16_t)hv;
      }
    __syncthreads();

    // ---- down: acc += h(32x128) @ Wdown[fc:fc+128, w*128 .. +128] ----
    for (int kk = 0; kk < FC; kk += 32) {
      v16bf ha[2];
#pragma unroll
      for (int mt = 0; mt < 2; mt++) {
        const bf16_t* hr = lds_h + (mt * 16 + ln) * HS + kk;
        ha[mt] = make_frag(hr + 16 * hi, hr + 16 * hi + 8);
      }
      const int kt = (fc + kk) >> 5;
#pragma unroll
      for (int nt = 0; nt < 8; nt++) {
        const int col = w * 128 + nt * 16;
        const bf16_t* pd = wdP + ((size_t)(col >> 4) * (F_DIM / 32) + kt) * 512
                               + lane * 16;
        const v16bf bd = make_frag(pd, pd + 8);
        acc[0][nt] = wmma_bf16(ha[0], bd, acc[0][nt]);
        acc[1][nt] = wmma_bf16(ha[1], bd, acc[1][nt]);
      }
    }
    __syncthreads();
  }

  // ---- gate-scaled scatter ----
#pragma unroll
  for (int nt = 0; nt < 8; nt++) {
    const int col = w * 128 + nt * 16 + ln;
#pragma unroll
    for (int mt = 0; mt < 2; mt++)
#pragma unroll
      for (int v = 0; v < 8; v++) {
        const int m = mt * 16 + v + 8 * hi;
        const int t = s_tok[m];
        if (t >= 0)
          atomicAdd(out + (size_t)t * D_DIM + col, acc[mt][nt][v] * s_gate[m]);
      }
  }
}

// ---------------------------------------------------------------------------
// Kernel 3 (FALLBACK, small workspace): M=16, per-wave LDS staging of f32
// weights (round-1 kernel with fast silu).
// ---------------------------------------------------------------------------
__global__ __launch_bounds__(256) void moe_expert_kernel(
    const float* __restrict__ x,
    const float* __restrict__ Wup,
    const float* __restrict__ Wgate,
    const float* __restrict__ Wdown,
    const unsigned int* __restrict__ cnt,
    const int* __restrict__ toklist,
    const float* __restrict__ gatelist,
    float* __restrict__ out) {
  const int e    = blockIdx.y;
  const unsigned n_e = cnt[e];
  const int row0 = blockIdx.x * 16;
  if ((unsigned)row0 >= n_e) return;

  const int tid  = threadIdx.x;
  const int lane = tid & 31;
  const int w    = tid >> 5;
  const int hi   = lane >> 4;
  const int ln   = lane & 15;

  __shared__ int    s_tok[16];
  __shared__ float  s_gate[16];
  __shared__ __align__(16) bf16_t lds_x[16 * XS];
  __shared__ __align__(16) bf16_t lds_h[16 * HS];
  __shared__ __align__(16) bf16_t lds_w[8][2 * 16 * WSLOT];

  if (tid < 16) {
    const int r = row0 + tid;
    const bool v = (unsigned)r < n_e;
    s_tok[tid]  = v ? toklist [(size_t)e * T_TOK + r] : -1;
    s_gate[tid] = v ? gatelist[(size_t)e * T_TOK + r] : 0.0f;
  }
  __syncthreads();

  for (int idx = tid; idx < 16 * (D_DIM / 4); idx += 256) {
    const int m  = idx >> 8;
    const int d4 = (idx & 255) << 2;
    int t = s_tok[m]; if (t < 0) t = 0;
    const float4 v = *(const float4*)(x + (size_t)t * D_DIM + d4);
    v4bf b; b[0] = (bf16_t)v.x; b[1] = (bf16_t)v.y;
            b[2] = (bf16_t)v.z; b[3] = (bf16_t)v.w;
    *(v4bf*)(lds_x + m * XS + d4) = b;
  }
  __syncthreads();

  const float* wu = Wup   + (size_t)e * D_DIM * F_DIM;
  const float* wg = Wgate + (size_t)e * D_DIM * F_DIM;
  const float* wd = Wdown + (size_t)e * F_DIM * D_DIM;

  v8f acc[8];
#pragma unroll
  for (int i = 0; i < 8; i++) acc[i] = v8f{};

  bf16_t* slotU = lds_w[w];
  bf16_t* slotG = lds_w[w] + 16 * WSLOT;

  for (int fc = 0; fc < F_DIM; fc += FC) {
    const int nb = fc + w * 16;
    v8f uacc = v8f{}, gacc = v8f{};

    for (int kk = 0; kk < D_DIM; kk += 32) {
      const float* gu = wu + (size_t)kk * F_DIM + nb;
      const float* gg = wg + (size_t)kk * F_DIM + nb;
      for (int idx = lane; idx < 512; idx += 32) {
        const int k = idx >> 4, n = idx & 15;
        slotU[n * WSLOT + k] = (bf16_t)gu[(size_t)k * F_DIM + n];
        slotG[n * WSLOT + k] = (bf16_t)gg[(size_t)k * F_DIM + n];
      }
      const bf16_t* xr = lds_x + ln * XS + kk;
      const v16bf a = make_frag(xr + 16 * hi, xr + 16 * hi + 8);
      const bf16_t* bu = slotU + ln * WSLOT + 16 * hi;
      const bf16_t* bg = slotG + ln * WSLOT + 16 * hi;
      uacc = wmma_bf16(a, make_frag(bu, bu + 8), uacc);
      gacc = wmma_bf16(a, make_frag(bg, bg + 8), gacc);
    }

#pragma unroll
    for (int v = 0; v < 8; v++) {
      const float hv = silu_fast(uacc[v]) * gacc[v];
      lds_h[(v + 8 * hi) * HS + w * 16 + ln] = (bf16_t)hv;
    }
    __syncthreads();

    for (int kk = 0; kk < FC; kk += 32) {
      const bf16_t* hr = lds_h + ln * HS + kk;
      const v16bf a = make_frag(hr + 16 * hi, hr + 16 * hi + 8);
#pragma unroll
      for (int nt = 0; nt < 8; nt++) {
        const float* gd = wd + (size_t)(fc + kk) * D_DIM + (w * 128 + nt * 16);
        for (int idx = lane; idx < 512; idx += 32) {
          const int k = idx >> 4, n = idx & 15;
          slotU[n * WSLOT + k] = (bf16_t)gd[(size_t)k * D_DIM + n];
        }
        const bf16_t* bd = slotU + ln * WSLOT + 16 * hi;
        acc[nt] = wmma_bf16(a, make_frag(bd, bd + 8), acc[nt]);
      }
    }
    __syncthreads();
  }

#pragma unroll
  for (int nt = 0; nt < 8; nt++) {
    const int col = w * 128 + nt * 16 + ln;
#pragma unroll
    for (int v = 0; v < 8; v++) {
      const int m = v + 8 * hi;
      const int t = s_tok[m];
      if (t >= 0)
        atomicAdd(out + (size_t)t * D_DIM + col, acc[nt][v] * s_gate[m]);
    }
  }
}

// ---------------------------------------------------------------------------
// Host launcher
// ---------------------------------------------------------------------------
extern "C" void kernel_launch(void* const* d_in, const int* in_sizes, int n_in,
                              void* d_out, int out_size, void* d_ws, size_t ws_size,
                              hipStream_t stream) {
  (void)in_sizes; (void)n_in; (void)out_size;
  const float* x     = (const float*)d_in[0];
  const float* Wr    = (const float*)d_in[1];
  const float* Wup   = (const float*)d_in[2];
  const float* Wgate = (const float*)d_in[3];
  const float* Wdown = (const float*)d_in[4];
  float* out = (float*)d_out;

  unsigned char* ws = (unsigned char*)d_ws;
  unsigned int* cnt = (unsigned int*)ws;                    // 8 u32
  float* z_sum      = (float*)(ws + 32);
  float* p_sum      = (float*)(ws + 64);
  int*   toklist    = (int*)(ws + 128);                     // E*T ints
  float* gatelist   = (float*)(ws + 128 + (size_t)E_NUM * T_TOK * 4);

  const size_t PK_OFF = (size_t)1 << 20;                    // packed weights @ 1MB
  const size_t PK_SZ  = (size_t)E_NUM * D_DIM * F_DIM * sizeof(bf16_t); // 64MB
  bf16_t* pkUp   = (bf16_t*)(ws + PK_OFF);
  bf16_t* pkGate = (bf16_t*)(ws + PK_OFF + PK_SZ);
  bf16_t* pkDn   = (bf16_t*)(ws + PK_OFF + 2 * PK_SZ);
  const bool fast = ws_size >= PK_OFF + 3 * PK_SZ;

  const size_t out_elems = (size_t)T_TOK * D_DIM;
  moe_init_kernel<<<2048, 256, 0, stream>>>(out, out_elems, cnt);
  moe_router_kernel<<<T_TOK / 8, 256, 0, stream>>>(x, Wr, cnt, z_sum, p_sum,
                                                   toklist, gatelist);
  moe_finalize_kernel<<<1, 32, 0, stream>>>(cnt, z_sum, p_sum, out + out_elems);

  if (fast) {
    const int packBlocks = (int)((size_t)E_NUM * D_DIM * F_DIM / 16 / 256);
    moe_pack_kernel<<<packBlocks, 256, 0, stream>>>(Wup,   pkUp,   D_DIM, F_DIM);
    moe_pack_kernel<<<packBlocks, 256, 0, stream>>>(Wgate, pkGate, D_DIM, F_DIM);
    moe_pack_kernel<<<packBlocks, 256, 0, stream>>>(Wdown, pkDn,   F_DIM, D_DIM);
    moe_expert_fast_kernel<<<dim3(T_TOK / 32, E_NUM), 256, 0, stream>>>(
        x, pkUp, pkGate, pkDn, cnt, toklist, gatelist, out);
  } else {
    moe_expert_kernel<<<dim3(T_TOK / 16, E_NUM), 256, 0, stream>>>(
        x, Wup, Wgate, Wdown, cnt, toklist, gatelist, out);
  }
}
